// GNCAClassifier_45045617000624
// MI455X (gfx1250) — matrix operand order repaired and checked by reading, not compiled, not run on hardware
//
#include <hip/hip_runtime.h>

#define NN 50000
#define EE 800000
#define BB 16
#define INDIM 64
#define ED 8
#define HD 128
#define SD 128
#define NC 8
#define TSTEPS 12
#define KMSG 160   // 136 padded to a multiple of 32
#define MT 64      // edges per message block (4 row tiles of 16)

typedef __attribute__((ext_vector_type(16))) __bf16 v16bf;
typedef __attribute__((ext_vector_type(8)))  float  v8f;
typedef __attribute__((ext_vector_type(8)))  unsigned short ush8;
typedef __attribute__((ext_vector_type(16))) unsigned short ush16;

// fp32 -> bf16 bits, round-to-nearest-even
__device__ __forceinline__ unsigned short f2bf(float f) {
  unsigned u = __builtin_bit_cast(unsigned, f);
  unsigned r = u + 0x7FFFu + ((u >> 16) & 1u);
  return (unsigned short)(r >> 16);
}

union AFrag { v16bf v; ush8 h[2]; };
union BFrag { v16bf v; ush8 h[2]; };

// A fragment (16x32 bf16 tile at k0) from an LDS row-major tile with row stride KP.
__device__ __forceinline__ v16bf load_a_lds(const unsigned short* lds, int KP, int k0, int lane) {
  int r = lane & 15, half = lane >> 4;
  AFrag a;
  a.h[0] = *(const ush8*)(lds + r * KP + k0 + half * 8);
  a.h[1] = *(const ush8*)(lds + r * KP + k0 + 16 + half * 8);
  return a.v;
}

// B fragment (32x16 bf16 at k0) from a pre-transposed weight wT[N][KP] (K contiguous).
__device__ __forceinline__ v16bf load_b_glb(const unsigned short* __restrict__ wT,
                                            int KP, int k0, int colbase, int lane) {
  int c = colbase + (lane & 15), half = lane >> 4;
  BFrag b;
  b.h[0] = *(const ush8*)(wT + c * KP + k0 + half * 8);
  b.h[1] = *(const ush8*)(wT + c * KP + k0 + 16 + half * 8);
  return b.v;
}

#define WMMA_BF16(acc, a, b) \
  __builtin_amdgcn_wmma_f32_16x16x32_bf16(false, (a), false, (b), (short)0, (acc), false, false)

// ---- CDNA5 async gather: global -> LDS, per-lane addresses, tracked on ASYNCcnt ----
// LDS byte offset = low 32 bits of the generic shared-space address.
__device__ __forceinline__ void async_g2l_b128(void* ldsdst, const void* gsrc) {
  unsigned l = (unsigned)(unsigned long long)ldsdst;
  unsigned long long g = (unsigned long long)gsrc;
  asm volatile("global_load_async_to_lds_b128 %0, %1, off" :: "v"(l), "v"(g) : "memory");
}
__device__ __forceinline__ void wait_async0() {
  asm volatile("s_wait_asynccnt 0" ::: "memory");
}

// ---------------- prep kernels ----------------

__global__ __launch_bounds__(256) void prep_weights_kernel(
    const float* __restrict__ enc_w, const float* __restrict__ msg_w,
    const float* __restrict__ up_w1, const float* __restrict__ up_w2,
    unsigned short* __restrict__ enc_wT, unsigned short* __restrict__ msg_wT,
    unsigned short* __restrict__ up_w1T, unsigned short* __restrict__ up_w2T) {
  int i = blockIdx.x * 256 + threadIdx.x;
  const int N0 = SD * INDIM;
  const int N1 = SD * KMSG;
  const int N2 = SD * (SD + HD);
  const int N3 = SD * HD;
  if (i < N0) {
    int n = i / INDIM, k = i % INDIM;
    enc_wT[i] = f2bf(enc_w[k * SD + n]);
  } else if (i < N0 + N1) {
    int j = i - N0; int n = j / KMSG, k = j % KMSG;
    msg_wT[j] = (k < SD + ED) ? f2bf(msg_w[k * HD + n]) : (unsigned short)0;
  } else if (i < N0 + N1 + N2) {
    int j = i - N0 - N1; int n = j / (SD + HD), k = j % (SD + HD);
    up_w1T[j] = f2bf(up_w1[k * HD + n]);
  } else if (i < N0 + N1 + N2 + N3) {
    int j = i - N0 - N1 - N2; int n = j / HD, k = j % HD;
    up_w2T[j] = f2bf(up_w2[k * SD + n]);
  }
}

__global__ __launch_bounds__(256) void ea_convert_kernel(
    const float* __restrict__ ea, unsigned short* __restrict__ ea_bf) {
  long long i = (long long)blockIdx.x * 256 + threadIdx.x;
  if (i < (long long)EE * ED) ea_bf[i] = f2bf(ea[i]);
}

__global__ __launch_bounds__(256) void zero_kernel(
    float* __restrict__ agg, float* __restrict__ deg) {
  long long i = (long long)blockIdx.x * 256 + threadIdx.x;
  if (i < (long long)NN * SD) agg[i] = 0.0f;
  if (i < NN) deg[i] = 0.0f;
}

__global__ __launch_bounds__(256) void deg_kernel(
    const long long* __restrict__ dst, float* __restrict__ deg) {
  int e = blockIdx.x * 256 + threadIdx.x;
  if (e < EE) atomicAdd(&deg[(int)dst[e]], 1.0f);
}

__global__ __launch_bounds__(256) void deginv_kernel(
    const float* __restrict__ deg, float* __restrict__ deg_inv) {
  int n = blockIdx.x * 256 + threadIdx.x;
  if (n < NN) {
    float d = deg[n];
    deg_inv[n] = 1.0f / (d > 1.0f ? d : 1.0f);
  }
}

// ---------------- encoder: state = relu(x @ enc_w + enc_b) ----------------

__global__ __launch_bounds__(256) void encoder_kernel(
    const float* __restrict__ x, const unsigned short* __restrict__ enc_wT,
    const float* __restrict__ enc_b, float* __restrict__ state,
    unsigned short* __restrict__ state_bf) {
  __shared__ __align__(32) unsigned short As[16 * INDIM];
  const int n0 = blockIdx.x * 16;
  for (int i = threadIdx.x; i < 16 * INDIM; i += 256) {
    int r = i >> 6, k = i & 63;
    As[i] = f2bf(x[(n0 + r) * INDIM + k]);
  }
  __syncthreads();
  const int lane = threadIdx.x & 31, wave = threadIdx.x >> 5;
  const int colbase = wave * 16;
  v8f acc = {};
#pragma unroll
  for (int k0 = 0; k0 < INDIM; k0 += 32) {
    v16bf a = load_a_lds(As, INDIM, k0, lane);
    v16bf b = load_b_glb(enc_wT, INDIM, k0, colbase, lane);
    acc = WMMA_BF16(acc, a, b);
  }
  const int col = colbase + (lane & 15), half = lane >> 4;
  const float bias = enc_b[col];
#pragma unroll
  for (int i = 0; i < 8; ++i) {
    int row = n0 + half * 8 + i;
    float v = acc[i] + bias;
    v = v > 0.0f ? v : 0.0f;
    state[row * SD + col] = v;
    state_bf[row * SD + col] = f2bf(v);
  }
}

// ------- message: m = relu([state[src], ea] @ msg_w + b); agg[dst] += m -------
// 64 edges per block; A gathered via async global->LDS; B held in registers
// across the 4 row tiles (20 WMMAs per wave).

__global__ __launch_bounds__(256) void message_kernel(
    const unsigned short* __restrict__ state_bf,
    const unsigned short* __restrict__ ea_bf,
    const long long* __restrict__ src, const long long* __restrict__ dst,
    const unsigned short* __restrict__ msg_wT, const float* __restrict__ msg_b,
    float* __restrict__ agg) {
  __shared__ __align__(32) unsigned short As[MT * KMSG];
  __shared__ int dsts[MT];
  const long long e0 = (long long)blockIdx.x * MT;
  const int tid = threadIdx.x;

  // async gather: each state row = 16 segments of 16B (b128)
#pragma unroll
  for (int t = tid; t < MT * 16; t += 256) {
    int r = t >> 4, seg = t & 15;
    int s = (int)src[e0 + r];
    async_g2l_b128(&As[r * KMSG + seg * 8], &state_bf[(long long)s * SD + seg * 8]);
  }
  // async edge_attr: one b128 per row
  if (tid < MT) {
    async_g2l_b128(&As[tid * KMSG + SD], &ea_bf[(e0 + tid) * ED]);
    dsts[tid] = (int)dst[e0 + tid];
  }
  // zero the K padding (elements 136..159 of each row = 12 dwords)
  {
    unsigned* As32 = (unsigned*)As;
    for (int t = tid; t < MT * 12; t += 256) {
      int r = t / 12, j = t % 12;
      As32[r * (KMSG / 2) + (SD + ED) / 2 + j] = 0u;
    }
  }
  wait_async0();
  __syncthreads();

  const int lane = tid & 31, wave = tid >> 5;
  const int colbase = wave * 16;
  v16bf Bf[KMSG / 32];
#pragma unroll
  for (int kc = 0; kc < KMSG / 32; ++kc)
    Bf[kc] = load_b_glb(msg_wT, KMSG, kc * 32, colbase, lane);

  const int col = colbase + (lane & 15), half = lane >> 4;
  const float bias = msg_b[col];
#pragma unroll
  for (int rt = 0; rt < MT / 16; ++rt) {
    const unsigned short* Atile = As + rt * 16 * KMSG;
    v8f acc = {};
#pragma unroll
    for (int kc = 0; kc < KMSG / 32; ++kc) {
      v16bf a = load_a_lds(Atile, KMSG, kc * 32, lane);
      acc = WMMA_BF16(acc, a, Bf[kc]);
    }
#pragma unroll
    for (int i = 0; i < 8; ++i) {
      int m = rt * 16 + half * 8 + i;
      float v = acc[i] + bias;
      v = v > 0.0f ? v : 0.0f;
      atomicAdd(&agg[dsts[m] * SD + col], v);
    }
  }
}

// --- update: h = relu([state, agg/deg] @ w1 + b1); state += h @ w2 + b2 ---

__global__ __launch_bounds__(256) void update_kernel(
    unsigned short* __restrict__ state_bf, float* __restrict__ state,
    float* __restrict__ agg, const float* __restrict__ deg_inv,
    const unsigned short* __restrict__ up_w1T, const float* __restrict__ up_b1,
    const unsigned short* __restrict__ up_w2T, const float* __restrict__ up_b2) {
  __shared__ __align__(32) unsigned short As[16 * (SD + HD)];
  __shared__ __align__(32) unsigned short Hs[16 * HD];
  const int n0 = blockIdx.x * 16;

  // state half of A: 16 rows x 16 b128-segments = exactly 256 async tasks
  {
    int t = threadIdx.x;
    int r = t >> 4, seg = t & 15;
    async_g2l_b128(&As[r * (SD + HD) + seg * 8],
                   &state_bf[(long long)(n0 + r) * SD + seg * 8]);
  }
  // agg half: scale by 1/deg, convert, and consume+reset agg for the next step
  for (int i = threadIdx.x; i < 16 * HD; i += 256) {
    int r = i >> 7, j = i & 127;
    int row = n0 + r;
    float a = agg[row * SD + j] * deg_inv[row];
    As[r * (SD + HD) + SD + j] = f2bf(a);
    agg[row * SD + j] = 0.0f;
  }
  wait_async0();
  __syncthreads();

  const int lane = threadIdx.x & 31, wave = threadIdx.x >> 5;
  const int colbase = wave * 16;
  const int col = colbase + (lane & 15), half = lane >> 4;

  v8f acc = {};
#pragma unroll
  for (int k0 = 0; k0 < SD + HD; k0 += 32) {
    v16bf a = load_a_lds(As, SD + HD, k0, lane);
    v16bf b = load_b_glb(up_w1T, SD + HD, k0, colbase, lane);
    acc = WMMA_BF16(acc, a, b);
  }
  const float b1 = up_b1[col];
#pragma unroll
  for (int i = 0; i < 8; ++i) {
    int m = half * 8 + i;
    float v = acc[i] + b1;
    v = v > 0.0f ? v : 0.0f;
    Hs[m * HD + col] = f2bf(v);
  }
  __syncthreads();

  v8f acc2 = {};
#pragma unroll
  for (int k0 = 0; k0 < HD; k0 += 32) {
    v16bf a = load_a_lds(Hs, HD, k0, lane);
    v16bf b = load_b_glb(up_w2T, HD, k0, colbase, lane);
    acc2 = WMMA_BF16(acc2, a, b);
  }
  const float b2 = up_b2[col];
#pragma unroll
  for (int i = 0; i < 8; ++i) {
    int row = n0 + half * 8 + i;
    float ns = state[row * SD + col] + acc2[i] + b2;
    state[row * SD + col] = ns;
    state_bf[row * SD + col] = f2bf(ns);
  }
}

// ---------------- readout ----------------

__global__ void out_zero_kernel(float* __restrict__ out, float* __restrict__ cnt) {
  int i = threadIdx.x;
  if (i < BB * NC) out[i] = 0.0f;
  if (i < BB) cnt[i] = 0.0f;
}

__global__ __launch_bounds__(256) void out_acc_kernel(
    const float* __restrict__ state, const long long* __restrict__ batch,
    float* __restrict__ out, float* __restrict__ cnt) {
  int n = blockIdx.x * 256 + threadIdx.x;
  if (n < NN) {
    int b = (int)batch[n];
    atomicAdd(&cnt[b], 1.0f);
#pragma unroll
    for (int c = 0; c < NC; ++c)
      atomicAdd(&out[b * NC + c], state[n * SD + (SD - NC) + c]);
  }
}

__global__ void out_fin_kernel(float* __restrict__ out, const float* __restrict__ cnt) {
  int i = threadIdx.x;
  if (i < BB * NC) {
    float c = cnt[i / NC];
    out[i] /= (c > 1.0f ? c : 1.0f);
  }
}

// ---------------- launch ----------------

extern "C" void kernel_launch(void* const* d_in, const int* in_sizes, int n_in,
                              void* d_out, int out_size, void* d_ws, size_t ws_size,
                              hipStream_t stream) {
  const float*     x     = (const float*)d_in[0];
  const long long* eidx  = (const long long*)d_in[1];
  const float*     ea    = (const float*)d_in[2];
  const long long* batch = (const long long*)d_in[3];
  const float*     enc_w = (const float*)d_in[4];
  const float*     enc_b = (const float*)d_in[5];
  const float*     msg_w = (const float*)d_in[6];
  const float*     msg_b = (const float*)d_in[7];
  const float*     up_w1 = (const float*)d_in[8];
  const float*     up_b1 = (const float*)d_in[9];
  const float*     up_w2 = (const float*)d_in[10];
  const float*     up_b2 = (const float*)d_in[11];
  const long long* src = eidx;
  const long long* dst = eidx + EE;

  char* ws = (char*)d_ws;
  size_t off = 0;
  auto alloc = [&](size_t bytes) -> void* {
    void* p = ws + off;
    off += (bytes + 255) & ~(size_t)255;
    return p;
  };
  float*          state    = (float*)alloc((size_t)NN * SD * 4);
  float*          agg      = (float*)alloc((size_t)NN * SD * 4);
  unsigned short* state_bf = (unsigned short*)alloc((size_t)NN * SD * 2);
  unsigned short* ea_bf    = (unsigned short*)alloc((size_t)EE * ED * 2);
  float*          deg      = (float*)alloc((size_t)NN * 4);
  float*          deg_inv  = (float*)alloc((size_t)NN * 4);
  unsigned short* enc_wT   = (unsigned short*)alloc((size_t)SD * INDIM * 2);
  unsigned short* msg_wT   = (unsigned short*)alloc((size_t)SD * KMSG * 2);
  unsigned short* up_w1T   = (unsigned short*)alloc((size_t)SD * (SD + HD) * 2);
  unsigned short* up_w2T   = (unsigned short*)alloc((size_t)SD * HD * 2);
  float*          cnt      = (float*)alloc((size_t)BB * 4);
  (void)ws_size; (void)in_sizes; (void)n_in; (void)out_size;

  const int wtot = SD * INDIM + SD * KMSG + SD * (SD + HD) + SD * HD;
  prep_weights_kernel<<<(wtot + 255) / 256, 256, 0, stream>>>(
      enc_w, msg_w, up_w1, up_w2, enc_wT, msg_wT, up_w1T, up_w2T);
  ea_convert_kernel<<<(int)(((long long)EE * ED + 255) / 256), 256, 0, stream>>>(ea, ea_bf);
  zero_kernel<<<(int)(((long long)NN * SD + 255) / 256), 256, 0, stream>>>(agg, deg);
  deg_kernel<<<(EE + 255) / 256, 256, 0, stream>>>(dst, deg);
  deginv_kernel<<<(NN + 255) / 256, 256, 0, stream>>>(deg, deg_inv);

  encoder_kernel<<<NN / 16, 256, 0, stream>>>(x, enc_wT, enc_b, state, state_bf);

  for (int t = 0; t < TSTEPS; ++t) {
    message_kernel<<<EE / MT, 256, 0, stream>>>(state_bf, ea_bf, src, dst, msg_wT, msg_b, agg);
    update_kernel<<<NN / 16, 256, 0, stream>>>(state_bf, state, agg, deg_inv,
                                               up_w1T, up_b1, up_w2T, up_b2);
  }

  out_zero_kernel<<<1, 128, 0, stream>>>((float*)d_out, cnt);
  out_acc_kernel<<<(NN + 255) / 256, 256, 0, stream>>>(state, batch, (float*)d_out, cnt);
  out_fin_kernel<<<1, 128, 0, stream>>>((float*)d_out, cnt);
}
